// RecurrentRetention_44495861186727
// MI455X (gfx1250) — matrix operand order repaired and checked by reading, not compiled, not run on hardware
//
#include <hip/hip_runtime.h>

typedef __bf16 bf16_t;
typedef __attribute__((ext_vector_type(16))) __bf16 v16bf;
typedef __attribute__((ext_vector_type(8)))  float  v8f;
typedef int v4i __attribute__((vector_size(16)));

constexpr int   Bn = 8, Tn = 2048, Dn = 1024;
constexpr int   Mn = Bn * Tn;                  // 16384 rows
constexpr float GAMMA = 0.9865f;
constexpr int   NCHUNK = 16, CT = Tn / NCHUNK; // 128 steps per chunk

#if defined(__gfx1250__) && __has_builtin(__builtin_amdgcn_global_load_async_to_lds_b128)
#define HAVE_ASYNC_LDS 1
#else
#define HAVE_ASYNC_LDS 0
#endif

// ---------------- fp32 -> bf16 bulk convert (8 elems / thread, b128 in/out) ---
__global__ __launch_bounds__(256)
void k_cvt_bf16(const float* __restrict__ in, bf16_t* __restrict__ out, int n) {
  int i = (blockIdx.x * 256 + threadIdx.x) * 8;
  if (i >= n) return;
  const float4* p = (const float4*)(in + i);
  float4 a = p[0], b = p[1];
  union { bf16_t h[8]; uint4 u; } r;
  r.h[0] = (bf16_t)a.x; r.h[1] = (bf16_t)a.y; r.h[2] = (bf16_t)a.z; r.h[3] = (bf16_t)a.w;
  r.h[4] = (bf16_t)b.x; r.h[5] = (bf16_t)b.y; r.h[6] = (bf16_t)b.z; r.h[7] = (bf16_t)b.w;
  *(uint4*)(out + i) = r.u;
}

// -------- fp32 -> bf16 convert + transpose: out[n*Dn+k] = bf16(in[k*Dn+n]) ----
__global__ __launch_bounds__(256)
void k_cvt_transpose(const float* __restrict__ in, bf16_t* __restrict__ out) {
  __shared__ bf16_t tile[32][33];                 // +1 pad: conflict-free
  int tx = threadIdx.x & 31;
  int ty = threadIdx.x >> 5;                      // 0..7
  int k0 = blockIdx.x * 32, n0 = blockIdx.y * 32;
#pragma unroll
  for (int i = 0; i < 32; i += 8)
    tile[ty + i][tx] = (bf16_t)in[(size_t)(k0 + ty + i) * Dn + n0 + tx];
  __syncthreads();
#pragma unroll
  for (int i = 0; i < 32; i += 8)
    out[(size_t)(n0 + ty + i) * Dn + k0 + tx] = tile[tx][ty + i];
}

// ---------------- wave32 reduction ------------------------------------------
__device__ __forceinline__ float wave_sum(float v) {
#pragma unroll
  for (int off = 16; off > 0; off >>= 1) v += __shfl_xor(v, off, 32);
  return v;
}

// wsum[d] = sum_e Wk[d][e]   (one wave per row)
__global__ __launch_bounds__(256)
void k_rowsum(const float* __restrict__ W, float* __restrict__ wsum) {
  int wid = threadIdx.x >> 5, lane = threadIdx.x & 31;
  int row = blockIdx.x * 8 + wid;
  const float* p = W + (size_t)row * Dn;
  float s = 0.f;
  for (int i = lane; i < Dn; i += 32) s += p[i];
  s = wave_sum(s);
  if (lane == 0) wsum[row] = s;
}

// ksum[b*T+t] = xk[row,:] . wsum   (one wave per row; replaces the whole K GEMM)
__global__ __launch_bounds__(256)
void k_ksum(const float* __restrict__ xk, const float* __restrict__ wsum,
            float* __restrict__ ksum) {
  int wid = threadIdx.x >> 5, lane = threadIdx.x & 31;
  int row = blockIdx.x * 8 + wid;
  const float* p = xk + (size_t)row * Dn;
  float s = 0.f;
  for (int i = lane; i < Dn; i += 32) s += p[i] * wsum[i];
  s = wave_sum(s);
  if (lane == 0) ksum[row] = s;
}

// ---------------- bf16 WMMA GEMM: C[M,N] = A[M,K] * Wt[N,K]^T ----------------
// Block 256 threads = 8 waves (2x4), tile 128x128, k-step 32.
// Wt is pre-transposed (row-major [n][k]) so A and B staging paths are identical.
union Frag { v16bf v; uint4 q[2]; };

__device__ __forceinline__ void wmma_tile(const bf16_t* __restrict__ As,
                                          const bf16_t* __restrict__ Bs,
                                          v8f acc[4][2], int wm, int wn,
                                          int half, int l16) {
  Frag af[4], bfm[2];
#pragma unroll
  for (int i = 0; i < 4; ++i) {
    // A frag (16x32): lane<16 -> M=l16, K0-7 & K16-23; lanes>=16 -> K8-15 & K24-31
    const bf16_t* ap = &As[(wm * 64 + i * 16 + l16) * 32 + half * 8];
    af[i].q[0] = *(const uint4*)ap;
    af[i].q[1] = *(const uint4*)(ap + 16);
  }
#pragma unroll
  for (int j = 0; j < 2; ++j) {
    // B frag (32x16): lane<16 -> N=l16, K0-7 & K16-23; lanes>=16 -> K8-15 & K24-31
    const bf16_t* bp = &Bs[(wn * 32 + j * 16 + l16) * 32 + half * 8];
    bfm[j].q[0] = *(const uint4*)bp;
    bfm[j].q[1] = *(const uint4*)(bp + 16);
  }
#pragma unroll
  for (int i = 0; i < 4; ++i)
#pragma unroll
    for (int j = 0; j < 2; ++j)
      acc[i][j] = __builtin_amdgcn_wmma_f32_16x16x32_bf16(
          false, af[i].v, false, bfm[j].v, (short)0, acc[i][j], false, false);
}

#if HAVE_ASYNC_LDS
__device__ __forceinline__ void async_copy16(const bf16_t* g, bf16_t* l) {
  __builtin_amdgcn_global_load_async_to_lds_b128(
      (__attribute__((address_space(1))) v4i*)g,
      (__attribute__((address_space(3))) v4i*)l, 0, 0);
}
#endif

__global__ __launch_bounds__(256)
void k_gemm_bf16(const bf16_t* __restrict__ A, const bf16_t* __restrict__ Wt,
                 float* __restrict__ C) {
  const int tid  = threadIdx.x;
  const int lane = tid & 31, wid = tid >> 5;
  const int wm = wid >> 2, wn = wid & 3;          // wave grid 2 (M) x 4 (N)
  const int m0 = blockIdx.x * 128, n0 = blockIdx.y * 128;
  const int half = lane >> 4, l16 = lane & 15;
  const int ar = tid >> 1, ac = (tid & 1) * 16;   // 2 threads per 128-row, 16 k each

  v8f acc[4][2] = {};

#if HAVE_ASYNC_LDS
  __shared__ bf16_t As[2][128 * 32];              // double-buffered, 4 x 8 KiB
  __shared__ bf16_t Bs[2][128 * 32];
  const bf16_t* ag = A  + (size_t)(m0 + ar) * Dn + ac;
  const bf16_t* bg = Wt + (size_t)(n0 + ar) * Dn + ac;
  bf16_t* asl[2] = { &As[0][ar * 32 + ac], &As[1][ar * 32 + ac] };
  bf16_t* bsl[2] = { &Bs[0][ar * 32 + ac], &Bs[1][ar * 32 + ac] };

  auto issue = [&](int k0, int buf) {
    async_copy16(ag + k0,     asl[buf]);
    async_copy16(ag + k0 + 8, asl[buf] + 8);
    async_copy16(bg + k0,     bsl[buf]);
    async_copy16(bg + k0 + 8, bsl[buf] + 8);
  };

  issue(0, 0);
  int buf = 0;
  for (int k0 = 0; k0 < Dn - 32; k0 += 32) {
    issue(k0 + 32, buf ^ 1);                      // prefetch next tile
    asm volatile("s_wait_asynccnt 0x4" ::: "memory");  // current tile landed
    __syncthreads();
    wmma_tile(As[buf], Bs[buf], acc, wm, wn, half, l16);
    __syncthreads();                              // safe to overwrite this buffer
    buf ^= 1;
  }
  asm volatile("s_wait_asynccnt 0x0" ::: "memory");
  __syncthreads();
  wmma_tile(As[buf], Bs[buf], acc, wm, wn, half, l16);
#else
  __shared__ bf16_t As1[128 * 32];
  __shared__ bf16_t Bs1[128 * 32];
  for (int k0 = 0; k0 < Dn; k0 += 32) {
    const bf16_t* ag = A  + (size_t)(m0 + ar) * Dn + k0 + ac;
    const bf16_t* bg = Wt + (size_t)(n0 + ar) * Dn + k0 + ac;
    uint4 a0 = *(const uint4*)ag;
    uint4 a1 = *(const uint4*)(ag + 8);
    uint4 b0 = *(const uint4*)bg;
    uint4 b1 = *(const uint4*)(bg + 8);
    *(uint4*)&As1[ar * 32 + ac]     = a0;
    *(uint4*)&As1[ar * 32 + ac + 8] = a1;
    *(uint4*)&Bs1[ar * 32 + ac]     = b0;
    *(uint4*)&Bs1[ar * 32 + ac + 8] = b1;
    __syncthreads();
    wmma_tile(As1, Bs1, acc, wm, wn, half, l16);
    __syncthreads();
  }
#endif

  // C layout: lane half picks M 0-7 vs 8-15 per VGPR r, N = l16
#pragma unroll
  for (int i = 0; i < 4; ++i)
#pragma unroll
    for (int j = 0; j < 2; ++j) {
      int mb = m0 + wm * 64 + i * 16 + half * 8;
      int nn = n0 + wn * 32 + j * 16 + l16;
#pragma unroll
      for (int r = 0; r < 8; ++r)
        C[(size_t)(mb + r) * Dn + nn] = acc[i][j][r];
    }
}

// ---------------- chunked scan: s_t = g*s_{t-1} + ksum_t * V_t ---------------
__global__ __launch_bounds__(256)
void k_scan_chunks(const float* __restrict__ V, const float* __restrict__ ksum,
                   float* __restrict__ csum) {
  int idx = blockIdx.x * 256 + threadIdx.x;   // ((b*16+c)<<10) | d
  int d = idx & (Dn - 1);
  int bc = idx >> 10;
  int c = bc & (NCHUNK - 1), b = bc >> 4;
  int t0 = c * CT;
  const float* vp = V + ((size_t)(b * Tn + t0)) * Dn + d;
  const float* kp = ksum + b * Tn + t0;
  float s = 0.f;
  for (int t = 0; t < CT; ++t) {
    float u = ((t0 + t) > 0) ? kp[t] * vp[(size_t)t * Dn] : 0.f;
    s = s * GAMMA + u;
  }
  csum[idx] = s;
}

__global__ __launch_bounds__(256)
void k_scan_carry(const float* __restrict__ csum, float* __restrict__ carry) {
  int idx = blockIdx.x * 256 + threadIdx.x;   // b*Dn + d
  int d = idx & (Dn - 1), b = idx >> 10;
  float gC = 1.f;
#pragma unroll
  for (int i = 0; i < CT; ++i) gC *= GAMMA;   // gamma^128
  float s = 0.f;
  for (int c = 0; c < NCHUNK; ++c) {
    int o = ((b * NCHUNK + c) << 10) | d;
    carry[o] = s;
    s = s * gC + csum[o];
  }
}

__global__ __launch_bounds__(256)
void k_scan_apply(const float* __restrict__ V, const float* __restrict__ ksum,
                  const float* __restrict__ carry, float* __restrict__ out) {
  int idx = blockIdx.x * 256 + threadIdx.x;
  int d = idx & (Dn - 1);
  int bc = idx >> 10;
  int c = bc & (NCHUNK - 1), b = bc >> 4;
  int t0 = c * CT;
  size_t base = ((size_t)(b * Tn + t0)) * Dn + d;
  const float* kp = ksum + b * Tn + t0;
  float s = carry[idx];
  for (int t = 0; t < CT; ++t) {
    float u = ((t0 + t) > 0) ? kp[t] * V[base + (size_t)t * Dn] : 0.f;
    s = s * GAMMA + u;
    out[base + (size_t)t * Dn] *= s;          // out holds Q
  }
}

// ---------------- launcher ---------------------------------------------------
extern "C" void kernel_launch(void* const* d_in, const int* in_sizes, int n_in,
                              void* d_out, int out_size, void* d_ws, size_t ws_size,
                              hipStream_t stream) {
  const float* xq = (const float*)d_in[0];
  const float* xk = (const float*)d_in[1];
  const float* xv = (const float*)d_in[2];
  const float* Wq = (const float*)d_in[3];
  const float* Wk = (const float*)d_in[4];
  const float* Wv = (const float*)d_in[5];
  float* out = (float*)d_out;
  char*  ws  = (char*)d_ws;

  size_t off = 0;
  bf16_t* xq_b = (bf16_t*)(ws + off); off += (size_t)Mn * Dn * 2;   // 32 MiB
  bf16_t* xv_b = (bf16_t*)(ws + off); off += (size_t)Mn * Dn * 2;   // 32 MiB
  bf16_t* wq_t = (bf16_t*)(ws + off); off += (size_t)Dn * Dn * 2;   //  2 MiB (transposed)
  bf16_t* wv_t = (bf16_t*)(ws + off); off += (size_t)Dn * Dn * 2;   //  2 MiB (transposed)
  float*  Vf   = (float*)(ws + off);  off += (size_t)Mn * Dn * 4;   // 64 MiB
  float*  ksum = (float*)(ws + off);  off += (size_t)Mn * 4;
  float*  wsum = (float*)(ws + off);  off += (size_t)Dn * 4;
  float*  csum = (float*)(ws + off);  off += (size_t)Bn * NCHUNK * Dn * 4;
  float*  carry= (float*)(ws + off);  off += (size_t)Bn * NCHUNK * Dn * 4;

  const int nAct = Mn * Dn;   // 16M
  k_cvt_bf16<<<nAct / 2048, 256, 0, stream>>>(xq, xq_b, nAct);
  k_cvt_bf16<<<nAct / 2048, 256, 0, stream>>>(xv, xv_b, nAct);
  dim3 tg(Dn / 32, Dn / 32);
  k_cvt_transpose<<<tg, 256, 0, stream>>>(Wq, wq_t);
  k_cvt_transpose<<<tg, 256, 0, stream>>>(Wv, wv_t);
  k_rowsum<<<Dn / 8, 256, 0, stream>>>(Wk, wsum);
  k_ksum  <<<Mn / 8, 256, 0, stream>>>(xk, wsum, ksum);

  dim3 gg(Mn / 128, Dn / 128);                       // 128 x 8 blocks
  k_gemm_bf16<<<gg, 256, 0, stream>>>(xq_b, wq_t, out);  // Q straight into d_out
  k_gemm_bf16<<<gg, 256, 0, stream>>>(xv_b, wv_t, Vf);

  const int nsc = Bn * NCHUNK * Dn;                  // 131072 scan threads
  k_scan_chunks<<<nsc / 256, 256, 0, stream>>>(Vf, ksum, csum);
  k_scan_carry <<<(Bn * Dn) / 256, 256, 0, stream>>>(csum, carry);
  k_scan_apply <<<nsc / 256, 256, 0, stream>>>(Vf, ksum, carry, out);
}
